// RFFR_base_13975823582013
// MI455X (gfx1250) — matrix operand order repaired and checked by reading, not compiled, not run on hardware
//
#include <hip/hip_runtime.h>

typedef __attribute__((ext_vector_type(16))) _Float16 v16h;
typedef __attribute__((ext_vector_type(8)))  _Float16 v8h;
typedef __attribute__((ext_vector_type(2)))  _Float16 v2h;
typedef __attribute__((ext_vector_type(8)))  float    v8f;
typedef __attribute__((ext_vector_type(4)))  float    v4f;

#define B_  8
#define N_  4096
#define D_  512
#define BR  64     // output rows per workgroup
#define BM  64     // m-tile (sim columns / value rows) per iteration
#define NIT (N_ / BM)
#define QLD 520    // padded LDS row stride in halves for sQ / sK (512 + 8)
#define VLD 72     // padded LDS row stride in halves for sVt (64 + 8)
#define SLD 72     // padded LDS row stride in halves for sS  (64 + 8)

// ---------------- fragment loaders (wave32 WMMA f16 layouts) ----------------
// A-matrix 16x32 f16: lane = 16h + r holds row r; halves K = kb+8h..+7 and kb+16+8h..+7
__device__ __forceinline__ v16h ld_afrag(const _Float16* base, int ld, int r, int kb, int h) {
    const _Float16* p = base + r * ld + kb + 8 * h;
    v8h lo = *(const v8h*)(p);
    v8h hi = *(const v8h*)(p + 16);
    return __builtin_shufflevector(lo, hi, 0,1,2,3,4,5,6,7,8,9,10,11,12,13,14,15);
}
// B-matrix 32x16 f16: lane = 16h + n holds column n; halves K = kb+8h..+7 and kb+16+8h..+7
// storage is "NT": row n of the source matrix, contiguous along K
__device__ __forceinline__ v16h ld_bfrag(const _Float16* base, int ld, int n, int kb, int h) {
    const _Float16* p = base + n * ld + kb + 8 * h;
    v8h lo = *(const v8h*)(p);
    v8h hi = *(const v8h*)(p + 16);
    return __builtin_shufflevector(lo, hi, 0,1,2,3,4,5,6,7,8,9,10,11,12,13,14,15);
}

__device__ __forceinline__ float fast_tanh(float x) {
    // tanh(x) = (e^{2x}-1)/(e^{2x}+1); x is a cosine in [-1,1], no overflow risk
    float t = __builtin_amdgcn_exp2f(x * 2.8853900817779268f); // 2/ln2
    return (t - 1.0f) * __builtin_amdgcn_rcpf(t + 1.0f);
}

// ---------------- kernel 1: reciprocal row norms ----------------
__global__ __launch_bounds__(256) void rnorm_kernel(const float* __restrict__ tgt,
                                                    float* __restrict__ rnorm) {
    int row  = blockIdx.x * 8 + (threadIdx.x >> 5);   // one wave per row, B*N rows flat
    int lane = threadIdx.x & 31;
    const float* p = tgt + (size_t)row * D_;
    float s = 0.0f;
    #pragma unroll
    for (int k = 0; k < D_ / 32; ++k) {
        float v = p[lane + 32 * k];
        s += v * v;
    }
    #pragma unroll
    for (int off = 16; off > 0; off >>= 1)
        s += __shfl_xor(s, off, 32);
    if (lane == 0) rnorm[row] = rsqrtf(s);
}

// ---------------- kernel 2: fused  tanh(Qn Kn^T) @ V  ----------------
__global__ __launch_bounds__(256) void fused_kernel(const float* __restrict__ tgt,
                                                    const float* __restrict__ rnorm,
                                                    float* __restrict__ out) {
    extern __shared__ __align__(16) _Float16 smem[];
    _Float16* sQ  = smem;                         // BR x QLD  (normalized rows, f16)
    _Float16* sK  = sQ  + BR * QLD;               // BM x QLD  (normalized m-tile, f16)
    _Float16* sVt = sK  + BM * QLD;               // D_ x VLD  (raw m-tile, transposed)
    _Float16* sS  = sVt + (size_t)D_ * VLD;       // BR x SLD  (tanh(sim) tile, f16)

    const int tid  = threadIdx.x;
    const int w    = tid >> 5;        // wave 0..7
    const int lane = tid & 31;
    const int nl   = lane & 15;
    const int h    = lane >> 4;

    const int b  = blockIdx.y;
    const int n0 = blockIdx.x * BR;

    const float* tgtB = tgt + (size_t)b * N_ * D_;
    const float* rnB  = rnorm + (size_t)b * N_;

    // ---- stage Qn once: 4 threads per row, 128 cols each ----
    {
        int r  = tid >> 2;
        int c0 = (tid & 3) * 128;
        float rn = rnB[n0 + r];
        const float* src = tgtB + (size_t)(n0 + r) * D_ + c0;
        _Float16* dst = sQ + r * QLD + c0;
        #pragma unroll
        for (int k = 0; k < 128; k += 4) {
            v4f v = *(const v4f*)(src + k);
            dst[k + 0] = (_Float16)(v[0] * rn);
            dst[k + 1] = (_Float16)(v[1] * rn);
            dst[k + 2] = (_Float16)(v[2] * rn);
            dst[k + 3] = (_Float16)(v[3] * rn);
        }
    }

    // persistent accumulators: wave w owns cols [w*64, w*64+64), rows [0,64)
    v8f acc[4][4];
    #pragma unroll
    for (int i = 0; i < 4; ++i)
        #pragma unroll
        for (int j = 0; j < 4; ++j)
            #pragma unroll
            for (int e = 0; e < 8; ++e)
                acc[i][j][e] = 0.0f;

    const int si  = w >> 1;        // this wave's S row-tile (0..3)
    const int sj2 = (w & 1) * 2;   // this wave's S col-tile pair (0 or 2)

    for (int it = 0; it < NIT; ++it) {
        const int m0 = it * BM;
        __syncthreads();   // previous iteration done reading sK/sVt/sS

        // ---- stage Kn (normalized) + Vt (raw, transposed, b32-packed row pairs) ----
        {
            int p   = tid >> 3;           // row pair 0..31
            int c0  = (tid & 7) * 64;     // 64 cols per thread
            int r0  = m0 + 2 * p;
            float rn0 = rnB[r0];
            float rn1 = rnB[r0 + 1];
            const float* s0p = tgtB + (size_t)r0 * D_ + c0;
            const float* s1p = s0p + D_;
            _Float16* dk0 = sK + (2 * p) * QLD + c0;
            _Float16* dk1 = dk0 + QLD;
            #pragma unroll
            for (int k = 0; k < 64; k += 4) {
                v4f va = *(const v4f*)(s0p + k);
                v4f vb = *(const v4f*)(s1p + k);
                #pragma unroll
                for (int j = 0; j < 4; ++j) {
                    dk0[k + j] = (_Float16)(va[j] * rn0);
                    dk1[k + j] = (_Float16)(vb[j] * rn1);
                    v2h pk;
                    pk[0] = (_Float16)va[j];
                    pk[1] = (_Float16)vb[j];
                    *(v2h*)(sVt + (size_t)(c0 + k + j) * VLD + 2 * p) = pk;  // 4B store
                }
            }
        }
        __syncthreads();

        // ---- prefetch next m-tile into caches while we do matrix math ----
        if (it + 1 < NIT) {
            const float* np = tgtB + (size_t)(m0 + BM + (tid >> 2)) * D_ + (tid & 3) * 128;
            __builtin_prefetch(np +  0, 0, 0);
            __builtin_prefetch(np + 32, 0, 0);
            __builtin_prefetch(np + 64, 0, 0);
            __builtin_prefetch(np + 96, 0, 0);
        }

        // ---- GEMM1: two S tiles (si, sj2) (si, sj2+1) over D=512, pipelined ----
        v8f s0, s1;
        #pragma unroll
        for (int e = 0; e < 8; ++e) { s0[e] = 0.0f; s1[e] = 0.0f; }
        v16h a  = ld_afrag(sQ, QLD, 16 * si + nl, 0, h);
        v16h b0 = ld_bfrag(sK, QLD, 16 * sj2 + nl, 0, h);
        v16h b1 = ld_bfrag(sK, QLD, 16 * sj2 + 16 + nl, 0, h);
        #pragma unroll
        for (int kk = 0; kk < D_ / 32; ++kk) {
            v16h an = a, b0n = b0, b1n = b1;
            if (kk < D_ / 32 - 1) {          // prefetch next step's fragments
                an  = ld_afrag(sQ, QLD, 16 * si + nl, (kk + 1) * 32, h);
                b0n = ld_bfrag(sK, QLD, 16 * sj2 + nl, (kk + 1) * 32, h);
                b1n = ld_bfrag(sK, QLD, 16 * sj2 + 16 + nl, (kk + 1) * 32, h);
            }
            s0 = __builtin_amdgcn_wmma_f32_16x16x32_f16(false, a, false, b0,
                                                        (short)0, s0, false, false);
            s1 = __builtin_amdgcn_wmma_f32_16x16x32_f16(false, a, false, b1,
                                                        (short)0, s1, false, false);
            a = an; b0 = b0n; b1 = b1n;
        }

        // ---- tanh + store both S tiles as f16 ----
        #pragma unroll
        for (int e = 0; e < 8; ++e) {
            int row = 16 * si + e + 8 * h;
            sS[row * SLD + 16 * sj2      + nl] = (_Float16)fast_tanh(s0[e]);
            sS[row * SLD + 16 * sj2 + 16 + nl] = (_Float16)fast_tanh(s1[e]);
        }
        __syncthreads();

        // ---- GEMM2: acc[i][j] += S[16i..][0..64) x V[0..64)[w*64+16j..] ----
        #pragma unroll
        for (int ks = 0; ks < BM / 32; ++ks) {
            v16h bfr[4];
            #pragma unroll
            for (int j = 0; j < 4; ++j)
                bfr[j] = ld_bfrag(sVt, VLD, w * 64 + 16 * j + nl, 32 * ks, h);
            #pragma unroll
            for (int i = 0; i < 4; ++i) {
                v16h a2 = ld_afrag(sS, SLD, 16 * i + nl, 32 * ks, h);
                #pragma unroll
                for (int j = 0; j < 4; ++j)
                    acc[i][j] = __builtin_amdgcn_wmma_f32_16x16x32_f16(false, a2, false, bfr[j],
                                                                       (short)0, acc[i][j],
                                                                       false, false);
            }
        }
    }

    // ---- write output f32 (lanes 0..15 cover consecutive d -> coalesced b32) ----
    float* outB = out + ((size_t)b * N_ + n0) * D_;
    #pragma unroll
    for (int i = 0; i < 4; ++i)
        #pragma unroll
        for (int e = 0; e < 8; ++e) {
            int row = 16 * i + e + 8 * h;
            #pragma unroll
            for (int j = 0; j < 4; ++j) {
                int d = w * 64 + 16 * j + nl;
                outB[(size_t)row * D_ + d] = acc[i][j][e];
            }
        }
}

extern "C" void kernel_launch(void* const* d_in, const int* in_sizes, int n_in,
                              void* d_out, int out_size, void* d_ws, size_t ws_size,
                              hipStream_t stream) {
    const float* tgt = (const float*)d_in[0];
    float* out   = (float*)d_out;
    float* rnorm = (float*)d_ws;   // B*N floats = 128 KB scratch

    rnorm_kernel<<<B_ * N_ / 8, 256, 0, stream>>>(tgt, rnorm);

    size_t lds_bytes = (size_t)(BR * QLD + BM * QLD + (size_t)D_ * VLD + BR * SLD)
                       * sizeof(_Float16);   // ~211 KB (ISA allows up to 320 KB/WG)
    fused_kernel<<<dim3(N_ / BR, B_), 256, lds_bytes, stream>>>(tgt, rnorm, out);
}
